// ModernNCA_47528108097622
// MI455X (gfx1250) — matrix operand order repaired and checked by reading, not compiled, not run on hardware
//
#include <hip/hip_runtime.h>
#include <hip/hip_bf16.h>

typedef _Float16 v8h  __attribute__((ext_vector_type(8)));
typedef _Float16 v16h __attribute__((ext_vector_type(16)));
typedef float    v8f  __attribute__((ext_vector_type(8)));

// ---------------------------------------------------------------------------
// Problem constants
// ---------------------------------------------------------------------------
#define BROWS   512          // batch rows
#define NCAND   100000       // candidates
#define DIMK    128          // feature / encoded dim
#define NCLS    10           // classes
#define NTILESN (NCAND / 16) // 6250 candidate tiles
#define TILES_PER_CHUNK 32   // 512 candidates per chunk
#define NCHUNKS ((NTILESN + TILES_PER_CHUNK - 1) / TILES_PER_CHUNK) // 196
#define ACC_STRIDE 12        // 10 classes + denom + pad

// ---------------------------------------------------------------------------
// Helpers
// ---------------------------------------------------------------------------
static __device__ __forceinline__ v16h combine16(v8h lo, v8h hi) {
    return __builtin_shufflevector(lo, hi, 0,1,2,3,4,5,6,7,8,9,10,11,12,13,14,15);
}

static __device__ __forceinline__ v8f wmma_f16(v16h a, v16h b, v8f c) {
    // D = A(16xK=32 f16) * B(32x16 f16) + C(16x16 f32)
    return __builtin_amdgcn_wmma_f32_16x16x32_f16(
        /*neg_a=*/false, a, /*neg_b=*/false, b,
        /*c_mod=*/(short)0, c, /*reuse_a=*/false, /*reuse_b=*/false);
}

// Load A-operand (16x32 f16, row-major source, rows = M):
//   lane l holds row (l&15); halves 0..7 -> K = kb*32 + hi8 + 0..7
//                            halves 8..15 -> K = kb*32 + 16 + hi8 + 0..7
static __device__ __forceinline__ v16h load_A(const _Float16* row, int kb, int hi8) {
    v8h lo = *(const v8h*)(row + kb*32 + hi8);
    v8h hi = *(const v8h*)(row + kb*32 + 16 + hi8);
    return combine16(lo, hi);
}

// Load B-operand (32x16 f16): lane l = column n (l&15);
//   halves 0..7 -> K = kb*32 + hi16/2 + 0..7; halves 8..15 -> K = kb*32 + 16 + hi16/2 + 0..7
static __device__ __forceinline__ v16h load_B(const _Float16* col_row, int kb, int hi16) {
    v8h lo = *(const v8h*)(col_row + kb*32 + hi16/2);
    v8h hi = *(const v8h*)(col_row + kb*32 + 16 + hi16/2);
    return combine16(lo, hi);
}

// ---------------------------------------------------------------------------
// Kernel 0: convert W (128x128 f32, row-major [out_dim][in_dim]) to f16
// ---------------------------------------------------------------------------
__global__ void nca_cvt_w(const float* __restrict__ W, _Float16* __restrict__ W16) {
    int i = blockIdx.x * 256 + threadIdx.x;
    if (i < DIMK * DIMK) W16[i] = (_Float16)W[i];
}

// ---------------------------------------------------------------------------
// Kernel 1: encoder  H = X @ W^T + b   (row-major f16 out) + fp32 sq-norms
// One wave = one 16-row tile; 8 column tiles x 4 WMMAs each.
// ---------------------------------------------------------------------------
__global__ void nca_encode(const float* __restrict__ X,
                           const _Float16* __restrict__ W16,
                           const float* __restrict__ bias,
                           _Float16* __restrict__ H16,
                           float* __restrict__ Hn,
                           int ntiles) {
    const int lane = threadIdx.x & 31;
    const int wid  = threadIdx.x >> 5;
    const int tile = blockIdx.x * 8 + wid;
    if (tile >= ntiles) return;               // wave-uniform exit (EXEC stays all-1s)

    const int rowbase = tile * 16;
    const int m   = lane & 15;
    const int hi  = lane >> 4;
    const int hi8 = hi * 8;

    // A operands: rows of X, converted f32 -> f16 on the fly
    v16h A[4];
    const float* xrow = X + (size_t)(rowbase + m) * DIMK;
    #pragma unroll
    for (int kb = 0; kb < 4; ++kb) {
        union { v16h v; _Float16 e[16]; } u;
        const float* s0 = xrow + kb*32 + hi8;
        const float* s1 = xrow + kb*32 + 16 + hi8;
        #pragma unroll
        for (int j = 0; j < 8; ++j) {
            u.e[j]     = (_Float16)s0[j];
            u.e[8 + j] = (_Float16)s1[j];
        }
        A[kb] = u.v;
    }

    float nq[8];
    #pragma unroll
    for (int r = 0; r < 8; ++r) nq[r] = 0.0f;

    #pragma unroll
    for (int ct = 0; ct < 8; ++ct) {          // 8 output-dim tiles of 16
        v8f C = {0.f,0.f,0.f,0.f,0.f,0.f,0.f,0.f};
        const _Float16* wrow = W16 + (size_t)(ct*16 + m) * DIMK;  // B[k][n] = W[n][k]
        #pragma unroll
        for (int kb = 0; kb < 4; ++kb) {
            v16h B = load_B(wrow, kb, hi * 16);
            C = wmma_f16(A[kb], B, C);
        }
        const float bv = bias[ct*16 + m];
        #pragma unroll
        for (int r = 0; r < 8; ++r) {
            // C element (r,lane): row M = r + 8*hi, col N = m
            _Float16 hh = (_Float16)(C[r] + bv);
            H16[(size_t)(rowbase + r + hi8) * DIMK + ct*16 + m] = hh;
            float hf = (float)hh;             // norm consistent with f16 values
            nq[r] = fmaf(hf, hf, nq[r]);
        }
    }

    // reduce squared norms across the 16 lanes of each half
    #pragma unroll
    for (int r = 0; r < 8; ++r) {
        float v = nq[r];
        v += __shfl_xor(v, 1, 32);
        v += __shfl_xor(v, 2, 32);
        v += __shfl_xor(v, 4, 32);
        v += __shfl_xor(v, 8, 32);
        if (lane == 0)  Hn[rowbase + r]     = v;
        if (lane == 16) Hn[rowbase + 8 + r] = v;
    }
}

// ---------------------------------------------------------------------------
// Kernel 2: fused distance + exp + class accumulation.
// grid = (NCHUNKS, 4 row-blocks); 8 waves, each owns 16 h-rows.
// Scores s = -dist <= 0  =>  exp(s) in (0,1]; fp32 accumulation, no max shift.
// ---------------------------------------------------------------------------
__global__ void nca_main(const _Float16* __restrict__ H16,
                         const float* __restrict__ Hn,
                         const _Float16* __restrict__ HC16,
                         const float* __restrict__ HCn,
                         const int* __restrict__ Y,
                         float* __restrict__ part) {
    __shared__ float s_acc[128 * ACC_STRIDE];

    const int tid  = threadIdx.x;
    const int lane = tid & 31;
    const int wid  = tid >> 5;
    for (int i = tid; i < 128 * ACC_STRIDE; i += 256) s_acc[i] = 0.0f;
    __syncthreads();

    const int chunk   = blockIdx.x;
    const int rowblk  = blockIdx.y;
    const int rowbase = rowblk * 128 + wid * 16;
    const int m   = lane & 15;
    const int hi  = lane >> 4;
    const int hi8 = hi * 8;

    // A operands: this wave's 16 encoded rows (resident for the whole chunk)
    const _Float16* hrow = H16 + (size_t)(rowbase + m) * DIMK;
    v16h A[4];
    #pragma unroll
    for (int kb = 0; kb < 4; ++kb) A[kb] = load_A(hrow, kb, hi8);

    // per-(VGPR r, half) row squared norm, broadcast via bpermute
    const float hn_lane = Hn[rowbase + m];
    float hnv[8];
    #pragma unroll
    for (int r = 0; r < 8; ++r) hnv[r] = __shfl(hn_lane, r + hi8, 32);

    float dreg[8];
    #pragma unroll
    for (int r = 0; r < 8; ++r) dreg[r] = 0.0f;

    const int t0 = chunk * TILES_PER_CHUNK;
    const int t1 = (t0 + TILES_PER_CHUNK < NTILESN) ? t0 + TILES_PER_CHUNK : NTILESN;

    for (int t = t0; t < t1; ++t) {
        const int c0 = t * 16;
        const _Float16* crow = HC16 + (size_t)(c0 + m) * DIMK;  // B[k][n] = hc[c0+n][k]
        v8f S = {0.f,0.f,0.f,0.f,0.f,0.f,0.f,0.f};
        #pragma unroll
        for (int kb = 0; kb < 4; ++kb) {
            v16h B = load_B(crow, kb, hi * 16);
            S = wmma_f16(A[kb], B, S);
        }
        const float hcn_l = HCn[c0 + m];
        const int   yc    = Y[c0 + m];
        #pragma unroll
        for (int r = 0; r < 8; ++r) {
            float d2 = fmaf(S[r], -2.0f, hnv[r] + hcn_l);
            d2 = fmaxf(d2, 0.0f);
            float d = sqrtf(d2);
            float e = __expf(-d);
            dreg[r] += e;                                  // denominator partial
            atomicAdd(&s_acc[(wid*16 + r + hi8) * ACC_STRIDE + yc], e);  // ds_add_f32
        }
    }

    // denominators: reduce over the 16 lanes of each half, one writer per row
    #pragma unroll
    for (int r = 0; r < 8; ++r) {
        float v = dreg[r];
        v += __shfl_xor(v, 1, 32);
        v += __shfl_xor(v, 2, 32);
        v += __shfl_xor(v, 4, 32);
        v += __shfl_xor(v, 8, 32);
        if (lane == 0)  s_acc[(wid*16 + r)     * ACC_STRIDE + 10] = v;
        if (lane == 16) s_acc[(wid*16 + 8 + r) * ACC_STRIDE + 10] = v;
    }
    __syncthreads();

    // deterministic partials: part[row][chunk][12]
    for (int i = tid; i < 128 * ACC_STRIDE; i += 256) {
        int r = i / ACC_STRIDE, c = i % ACC_STRIDE;
        part[((size_t)(rowblk*128 + r) * NCHUNKS + chunk) * ACC_STRIDE + c] = s_acc[i];
    }
}

// ---------------------------------------------------------------------------
// Kernel 3: fixed-order chunk reduction + log(p + eps)
// ---------------------------------------------------------------------------
__global__ void nca_reduce(const float* __restrict__ part, float* __restrict__ out) {
    int row = blockIdx.x * blockDim.x + threadIdx.x;
    if (row >= BROWS) return;
    float s[ACC_STRIDE];
    #pragma unroll
    for (int j = 0; j < ACC_STRIDE; ++j) s[j] = 0.0f;
    const float* p = part + (size_t)row * NCHUNKS * ACC_STRIDE;
    for (int ch = 0; ch < NCHUNKS; ++ch) {
        #pragma unroll
        for (int j = 0; j < ACC_STRIDE; ++j) s[j] += p[ch * ACC_STRIDE + j];
    }
    const float den = s[10];
    #pragma unroll
    for (int c = 0; c < NCLS; ++c)
        out[row * NCLS + c] = logf(s[c] / den + 1e-7f);
}

// ---------------------------------------------------------------------------
// Host launcher (graph-capture safe: only kernel launches on `stream`)
// ---------------------------------------------------------------------------
extern "C" void kernel_launch(void* const* d_in, const int* in_sizes, int n_in,
                              void* d_out, int out_size, void* d_ws, size_t ws_size,
                              hipStream_t stream) {
    const float* x  = (const float*)d_in[0];   // [512,128]
    const float* cx = (const float*)d_in[1];   // [100000,128]
    const int*   cy = (const int*)d_in[2];     // [100000]
    const float* W  = (const float*)d_in[3];   // [128,128]
    const float* b  = (const float*)d_in[4];   // [128]
    float* out = (float*)d_out;                // [512,10]

    char* ws = (char*)d_ws;
    // workspace layout (all 16B-aligned):
    const size_t o_W16  = 0;                                  // 128*128*2   =    32768
    const size_t o_H16  = o_W16  + (size_t)DIMK*DIMK*2;       // 512*128*2   =   131072
    const size_t o_Hn   = o_H16  + (size_t)BROWS*DIMK*2;      // 512*4       =     2048
    const size_t o_HCn  = o_Hn   + (size_t)BROWS*4;           // 100000*4    =   400000
    const size_t o_HC16 = o_HCn  + (size_t)NCAND*4;           // 100000*128*2= 25600000
    const size_t o_part = o_HC16 + (size_t)NCAND*DIMK*2;      // 512*196*12*4=  4816896
    (void)ws_size; (void)in_sizes; (void)n_in; (void)out_size; // ~31 MB total

    _Float16* W16  = (_Float16*)(ws + o_W16);
    _Float16* H16  = (_Float16*)(ws + o_H16);
    float*    Hn   = (float*)   (ws + o_Hn);
    float*    HCn  = (float*)   (ws + o_HCn);
    _Float16* HC16 = (_Float16*)(ws + o_HC16);
    float*    part = (float*)   (ws + o_part);

    // 1) W -> f16
    nca_cvt_w<<<(DIMK*DIMK + 255)/256, 256, 0, stream>>>(W, W16);

    // 2) encode batch rows and candidate rows (WMMA f16)
    nca_encode<<<(BROWS/16 + 7)/8, 256, 0, stream>>>(x,  W16, b, H16,  Hn,  BROWS/16);
    nca_encode<<<(NTILESN + 7)/8,  256, 0, stream>>>(cx, W16, b, HC16, HCn, NTILESN);

    // 3) fused distance / exp / class accumulation
    dim3 grid(NCHUNKS, BROWS/128);
    nca_main<<<grid, 256, 0, stream>>>(H16, Hn, HC16, HCn, cy, part);

    // 4) deterministic reduction + log
    nca_reduce<<<(BROWS + 255)/256, 256, 0, stream>>>(part, out);
}
